// CrossAttentionBlockLLaMA_35278861370028
// MI455X (gfx1250) — matrix operand order repaired and checked by reading, not compiled, not run on hardware
//
#include <hip/hip_runtime.h>

#define B_ 2
#define S_ 2048
#define D_ 2048
#define H_ 16
#define HD_ 128
#define FF_ 5632
#define MROWS (B_*S_)   // 4096
#define BK 32

typedef __attribute__((ext_vector_type(16))) __bf16 bf16x16;
typedef __attribute__((ext_vector_type(8)))  __bf16 bf16x8;
typedef __attribute__((ext_vector_type(8)))  float  f32x8;
typedef __attribute__((ext_vector_type(4)))  unsigned int u32x4;
typedef __attribute__((ext_vector_type(8)))  int i32x8;
typedef __attribute__((ext_vector_type(4)))  int i32x4;

union FragAB { bf16x16 v; bf16x8 h[2]; };

__device__ __forceinline__ f32x8 wmma_bf16(const FragAB& a, const FragAB& b, f32x8 c) {
  return __builtin_amdgcn_wmma_f32_16x16x32_bf16(false, a.v, false, b.v, (short)0, c, false, false);
}

// Generic-pointer low 32 bits == LDS byte offset (ISA aperture layout).
__device__ __forceinline__ unsigned int lds_addr_of(const void* p) {
  return (unsigned int)(uintptr_t)p;
}

// ---------------------------------------------------------------------------
// Tensor Data Mover: async 2D tile load (rows x BK bf16, row stride K elems)
// from global into LDS.  D# per CDNA5 ISA ch.8: group0 = {count=1, lds_addr,
// global_addr, type=2}; group1 = {data_size=2B, tensor dims, tile dims,
// stride}.  Tracked by TENSORcnt.
// ---------------------------------------------------------------------------
__device__ __forceinline__ void tdm_load_2d(const __bf16* g, unsigned int lds,
                                            int rows, int K) {
  unsigned long long ga = (unsigned long long)(uintptr_t)g;
  u32x4 g0 = { 1u,                               // count=1 (user descriptor)
               lds,                              // lds_addr (bytes)
               (unsigned int)ga,                 // global_addr[31:0]
               (unsigned int)(ga >> 32) | (2u << 30) };  // addr[56:32] | type=2
  i32x8 g1;
  g1[0] = 0x10000;                               // data_size=1 -> 2 bytes
  g1[1] = (int)(((unsigned)K & 0xFFFFu) << 16);  // tensor_dim0[15:0]
  g1[2] = (K >> 16) & 0xFFFF;                    // tensor_dim0[31:16], dim1 lo=0
  g1[3] = 0x10 | (BK << 16);                     // tensor_dim1 hi (=1M rows), tile_dim0=32
  g1[4] = rows;                                  // tile_dim1, tile_dim2=0
  g1[5] = K;                                     // tensor_dim0_stride[31:0]
  g1[6] = 0;                                     // stride0 hi, stride1 lo
  g1[7] = 0;                                     // stride1 hi
  i32x4 gz = {0, 0, 0, 0};
#if defined(__clang_major__) && (__clang_major__ >= 23)
  i32x8 gz8 = {0, 0, 0, 0, 0, 0, 0, 0};
  __builtin_amdgcn_tensor_load_to_lds(g0, g1, gz, gz, gz8, 0);
#else
  __builtin_amdgcn_tensor_load_to_lds(g0, g1, gz, gz, 0);
#endif
}

// ---------------------------------------------------------------------------
// fp32 -> bf16 convert (weights)
// ---------------------------------------------------------------------------
__global__ void cvt_f32_bf16(const float* __restrict__ in, __bf16* __restrict__ out, size_t n) {
  size_t i = (size_t)blockIdx.x * blockDim.x + threadIdx.x;
  size_t st = (size_t)gridDim.x * blockDim.x;
  for (; i < n; i += st) out[i] = (__bf16)in[i];
}

// ---------------------------------------------------------------------------
// RMSNorm (fp32 in) -> bf16 out.  One block per row, D_=2048, 256 thr x 8 elems
// ---------------------------------------------------------------------------
__global__ __launch_bounds__(256) void rmsnorm_bf16_k(
    const float* __restrict__ x, const float* __restrict__ w, __bf16* __restrict__ out) {
  const int row = blockIdx.x, tid = threadIdx.x;
  const float* xr = x + (size_t)row * D_;
  float vals[8]; float ss = 0.f;
#pragma unroll
  for (int i = 0; i < 8; i++) { float v = xr[tid + i * 256]; vals[i] = v; ss += v * v; }
  __shared__ float red[256];
  red[tid] = ss; __syncthreads();
  for (int o = 128; o > 0; o >>= 1) { if (tid < o) red[tid] += red[tid + o]; __syncthreads(); }
  float rinv = rsqrtf(red[0] * (1.0f / D_) + 1e-5f);
#pragma unroll
  for (int i = 0; i < 8; i++) {
    int c = tid + i * 256;
    out[(size_t)row * D_ + c] = (__bf16)(vals[i] * rinv * w[c]);
  }
}

// ---------------------------------------------------------------------------
// Residual add + RMSNorm, fp32 out (final epilogue)
// ---------------------------------------------------------------------------
__global__ __launch_bounds__(256) void resid_rmsnorm_k(
    const float* __restrict__ ff, const float* __restrict__ resid,
    const float* __restrict__ w, float* __restrict__ out) {
  const int row = blockIdx.x, tid = threadIdx.x;
  const float* fr = ff + (size_t)row * D_;
  const float* rr = resid + (size_t)row * D_;
  float vals[8]; float ss = 0.f;
#pragma unroll
  for (int i = 0; i < 8; i++) {
    float v = fr[tid + i * 256] + rr[tid + i * 256];
    vals[i] = v; ss += v * v;
  }
  __shared__ float red[256];
  red[tid] = ss; __syncthreads();
  for (int o = 128; o > 0; o >>= 1) { if (tid < o) red[tid] += red[tid + o]; __syncthreads(); }
  float rinv = rsqrtf(red[0] * (1.0f / D_) + 1e-5f);
#pragma unroll
  for (int i = 0; i < 8; i++) {
    int c = tid + i * 256;
    out[(size_t)row * D_ + c] = vals[i] * rinv * w[c];
  }
}

// ---------------------------------------------------------------------------
// SwiGLU elementwise: out = silu(g1) * g3  (bf16 in/out)
// ---------------------------------------------------------------------------
__global__ void swiglu_k(const __bf16* __restrict__ g1, const __bf16* __restrict__ g3,
                         __bf16* __restrict__ out, size_t n) {
  size_t i = (size_t)blockIdx.x * blockDim.x + threadIdx.x;
  size_t st = (size_t)gridDim.x * blockDim.x;
  for (; i < n; i += st) {
    float a = (float)g1[i], b = (float)g3[i];
    float s = a / (1.f + __expf(-a));
    out[i] = (__bf16)(s * b);
  }
}

// ---------------------------------------------------------------------------
// GEMM: C[M,N] = A[M,K] * W[N,K]^T, row-major, bf16 in, fp32 accumulate.
// Block tile 128x128, 8 waves of 32x64 (8 WMMA / wave / K-step).
// TDM (tensor_load_to_lds) double-buffers the 128x32 A/B tiles into LDS so
// the DMA of tile i+1 overlaps the WMMAs of tile i; wave0 issues & drains
// TENSORcnt, the block barrier publishes the tiles.
// ---------------------------------------------------------------------------
__global__ __launch_bounds__(256) void gemm_bf16_nt(
    const __bf16* __restrict__ A, const __bf16* __restrict__ Bw,
    void* __restrict__ Cout, int M, int N, int K, int out_bf16) {
  __shared__ __align__(16) __bf16 sA[2][128 * BK];
  __shared__ __align__(16) __bf16 sB[2][128 * BK];
  const int tid = threadIdx.x;
  const int m0 = blockIdx.y * 128, n0 = blockIdx.x * 128;
  const int wave = tid >> 5, lane = tid & 31, lh = lane & 15, hb = lane >> 4;
  const int wm = wave >> 1, wn = wave & 1;
  f32x8 acc[2][4] = {};
  const __bf16* aTile = A + (size_t)m0 * K;
  const __bf16* bTile = Bw + (size_t)n0 * K;

  if (tid < 32) {  // prologue: DMA tile 0
    tdm_load_2d(aTile, lds_addr_of(&sA[0][0]), 128, K);
    tdm_load_2d(bTile, lds_addr_of(&sB[0][0]), 128, K);
  }
  const int nk = K / BK;
  for (int ki = 0; ki < nk; ki++) {
    const int buf = ki & 1;
    if (tid < 32) {
      if (ki + 1 < nk) {  // DMA next tile into the other buffer, keep 2 in flight
        tdm_load_2d(aTile + (size_t)(ki + 1) * BK, lds_addr_of(&sA[buf ^ 1][0]), 128, K);
        tdm_load_2d(bTile + (size_t)(ki + 1) * BK, lds_addr_of(&sB[buf ^ 1][0]), 128, K);
        __builtin_amdgcn_s_wait_tensorcnt(2);  // current tile's 2 loads done
      } else {
        __builtin_amdgcn_s_wait_tensorcnt(0);
      }
    }
    __syncthreads();
    FragAB af[2], bfr[4];
#pragma unroll
    for (int s = 0; s < 2; s++) {  // A frag: K base {0,8} by lane half, +{0,16}
      int ar = wm * 32 + s * 16 + lh;
      af[s].h[0] = *(const bf16x8*)&sA[buf][ar * BK + hb * 8];
      af[s].h[1] = *(const bf16x8*)&sA[buf][ar * BK + 16 + hb * 8];
    }
#pragma unroll
    for (int t = 0; t < 4; t++) {  // B frag: K base {0,8} by lane half, +{0,16}
      int br = wn * 64 + t * 16 + lh;
      bfr[t].h[0] = *(const bf16x8*)&sB[buf][br * BK + hb * 8];
      bfr[t].h[1] = *(const bf16x8*)&sB[buf][br * BK + 16 + hb * 8];
    }
#pragma unroll
    for (int s = 0; s < 2; s++)
#pragma unroll
      for (int t = 0; t < 4; t++)
        acc[s][t] = wmma_bf16(af[s], bfr[t], acc[s][t]);
    __syncthreads();  // all waves done reading buf before it is DMA-overwritten
  }
#pragma unroll
  for (int s = 0; s < 2; s++)
#pragma unroll
    for (int t = 0; t < 4; t++)
#pragma unroll
      for (int r = 0; r < 8; r++) {
        int row = m0 + wm * 32 + s * 16 + (hb ? r + 8 : r);
        int col = n0 + wn * 64 + t * 16 + lh;
        float v = acc[s][t][r];
        if (out_bf16) ((__bf16*)Cout)[(size_t)row * N + col] = (__bf16)v;
        else          ((float*)Cout)[(size_t)row * N + col] = v;
      }
}

// ---------------------------------------------------------------------------
// Flash attention, bf16 QKV [B,S,H*HD], bf16 out.  Block = (b,h,128 q rows),
// 8 waves each own 16 q rows; cooperative K tile + transposed V tile in LDS;
// online softmax using wave32 C-layout (lane-half == 16 cols of one row).
// ---------------------------------------------------------------------------
__global__ __launch_bounds__(256) void flash_attn_k(
    const __bf16* __restrict__ Q, const __bf16* __restrict__ Km,
    const __bf16* __restrict__ V, __bf16* __restrict__ O) {
  __shared__ __align__(16) __bf16 sK[32 * 128];    // [key][d]
  __shared__ __align__(16) __bf16 sVt[128 * 32];   // [d][key]
  __shared__ __align__(16) __bf16 sP[8 * 16 * 32]; // per-wave P staging
  const int tid = threadIdx.x, wave = tid >> 5, lane = tid & 31;
  const int lh = lane & 15, hb = lane >> 4;
  const int b = blockIdx.z, h = blockIdx.y, q0 = blockIdx.x * 128;
  const int qbase = q0 + wave * 16;
  const size_t rs = (size_t)H_ * HD_;

  FragAB aq[4];
  {
    const __bf16* qp = Q + ((size_t)b * S_ + qbase + lh) * rs + h * HD_ + hb * 8;
#pragma unroll
    for (int c = 0; c < 4; c++) {
      aq[c].h[0] = *(const bf16x8*)(qp + c * 32);
      aq[c].h[1] = *(const bf16x8*)(qp + c * 32 + 16);
    }
  }
  float m[8], l[8];
#pragma unroll
  for (int r = 0; r < 8; r++) { m[r] = -3.0e38f; l[r] = 0.f; }
  f32x8 oacc[8] = {};
  const float scale = 0.08838834764831845f;  // 1/sqrt(128)

  for (int kt = 0; kt < S_; kt += 32) {
#pragma unroll
    for (int i = 0; i < 2; i++) {  // cooperative K/V tile load (32 keys x 128 d)
      int id = tid + i * 256;
      int key = id >> 4, d8 = (id & 15) * 8;
      const size_t goff = ((size_t)b * S_ + kt + key) * rs + h * HD_ + d8;
      bf16x8 kv = *(const bf16x8*)(Km + goff);
      *(bf16x8*)&sK[key * 128 + d8] = kv;
      bf16x8 vv = *(const bf16x8*)(V + goff);
#pragma unroll
      for (int e = 0; e < 8; e++) sVt[(d8 + e) * 32 + key] = vv[e];
    }
    if (kt + 32 < S_)  // gfx1250 global_prefetch for next K tile
      __builtin_prefetch(Km + ((size_t)b * S_ + kt + 32 + (tid >> 3)) * rs + h * HD_ + (tid & 7) * 16, 0, 1);
    __syncthreads();

    // scores: two 16x16 tiles over 32 keys, K-loop over HD in 4 chunks
    f32x8 sc[2] = {};
#pragma unroll
    for (int j = 0; j < 2; j++)
#pragma unroll
      for (int c = 0; c < 4; c++) {
        FragAB bk;
        const __bf16* kp = &sK[(j * 16 + lh) * 128 + c * 32 + hb * 8];
        bk.h[0] = *(const bf16x8*)kp;
        bk.h[1] = *(const bf16x8*)(kp + 16);
        sc[j] = wmma_bf16(aq[c], bk, sc[j]);
      }

    // online softmax (per register-row; reductions within 16-lane half)
#pragma unroll
    for (int r = 0; r < 8; r++) {
      float s0 = sc[0][r] * scale, s1 = sc[1][r] * scale;
      float mx = fmaxf(s0, s1);
#pragma unroll
      for (int off = 8; off >= 1; off >>= 1) mx = fmaxf(mx, __shfl_xor(mx, off, 16));
      float mnew = fmaxf(m[r], mx);
      float corr = __expf(m[r] - mnew);
      float p0 = __expf(s0 - mnew), p1 = __expf(s1 - mnew);
      float ps = p0 + p1;
#pragma unroll
      for (int off = 8; off >= 1; off >>= 1) ps += __shfl_xor(ps, off, 16);
      l[r] = l[r] * corr + ps;
      m[r] = mnew;
#pragma unroll
      for (int t = 0; t < 8; t++) oacc[t][r] *= corr;
      sc[0][r] = p0; sc[1][r] = p1;
    }

    // stage P (C-layout -> LDS -> A-layout), per-wave private region
    __bf16* pb = &sP[wave * 512];
#pragma unroll
    for (int j = 0; j < 2; j++)
#pragma unroll
      for (int r = 0; r < 8; r++)
        pb[(hb ? r + 8 : r) * 32 + j * 16 + lh] = (__bf16)sc[j][r];
    FragAB pa;
    pa.h[0] = *(const bf16x8*)&pb[lh * 32 + hb * 8];
    pa.h[1] = *(const bf16x8*)&pb[lh * 32 + 16 + hb * 8];

    // O += P(16x32) * V(32x128), 8 N-chunks of 16
#pragma unroll
    for (int t = 0; t < 8; t++) {
      FragAB bv;
      const __bf16* vp = &sVt[(t * 16 + lh) * 32 + hb * 8];
      bv.h[0] = *(const bf16x8*)vp;
      bv.h[1] = *(const bf16x8*)(vp + 16);
      oacc[t] = wmma_bf16(pa, bv, oacc[t]);
    }
    __syncthreads();
  }

#pragma unroll
  for (int t = 0; t < 8; t++)
#pragma unroll
    for (int r = 0; r < 8; r++) {
      int qi = qbase + (hb ? r + 8 : r);
      O[((size_t)b * S_ + qi) * rs + h * HD_ + t * 16 + lh] = (__bf16)(oacc[t][r] / l[r]);
    }
}

// ---------------------------------------------------------------------------
// Host orchestration
// ---------------------------------------------------------------------------
static inline size_t align256(size_t x) { return (x + 255) & ~(size_t)255; }

extern "C" void kernel_launch(void* const* d_in, const int* in_sizes, int n_in,
                              void* d_out, int out_size, void* d_ws, size_t ws_size,
                              hipStream_t stream) {
  (void)in_sizes; (void)n_in; (void)out_size; (void)ws_size;
  const float* x = (const float*)d_in[0];
  const float* y = (const float*)d_in[1];
  const float* attn_norm_w = (const float*)d_in[2];
  const float* wq[2] = {(const float*)d_in[3],  (const float*)d_in[7]};
  const float* wk[2] = {(const float*)d_in[4],  (const float*)d_in[8]};
  const float* wv[2] = {(const float*)d_in[5],  (const float*)d_in[9]};
  const float* wo[2] = {(const float*)d_in[6],  (const float*)d_in[10]};
  const float* w1[2] = {(const float*)d_in[11], (const float*)d_in[15]};
  const float* w2[2] = {(const float*)d_in[12], (const float*)d_in[16]};
  const float* w3[2] = {(const float*)d_in[13], (const float*)d_in[17]};
  const float* fnorm[2] = {(const float*)d_in[14], (const float*)d_in[18]};

  char* p = (char*)d_ws;
  auto take = [&](size_t elems, size_t esz) { char* r = p; p += align256(elems * esz); return r; };

  __bf16 *wqb[2], *wkb[2], *wvb[2], *wob[2], *w1b[2], *w2b[2], *w3b[2];
  for (int s = 0; s < 2; s++) {
    wqb[s] = (__bf16*)take((size_t)D_ * D_, 2);
    wkb[s] = (__bf16*)take((size_t)D_ * D_, 2);
    wvb[s] = (__bf16*)take((size_t)D_ * D_, 2);
    wob[s] = (__bf16*)take((size_t)D_ * D_, 2);
    w1b[s] = (__bf16*)take((size_t)FF_ * D_, 2);
    w2b[s] = (__bf16*)take((size_t)D_ * FF_, 2);
    w3b[s] = (__bf16*)take((size_t)FF_ * D_, 2);
  }
  __bf16* xn  = (__bf16*)take((size_t)MROWS * D_, 2);
  __bf16* yn  = (__bf16*)take((size_t)MROWS * D_, 2);
  __bf16* qb  = (__bf16*)take((size_t)MROWS * D_, 2);
  __bf16* kb  = (__bf16*)take((size_t)MROWS * D_, 2);
  __bf16* vb  = (__bf16*)take((size_t)MROWS * D_, 2);
  __bf16* aob = (__bf16*)take((size_t)MROWS * D_, 2);
  __bf16* hbf = (__bf16*)take((size_t)MROWS * D_, 2);
  __bf16* g1b = (__bf16*)take((size_t)MROWS * FF_, 2);
  __bf16* g3b = (__bf16*)take((size_t)MROWS * FF_, 2);
  __bf16* ffb = (__bf16*)take((size_t)MROWS * FF_, 2);
  float*  fff = (float*)take((size_t)MROWS * D_, 4);

  auto cvt = [&](const float* src, __bf16* dst, size_t n) {
    cvt_f32_bf16<<<1024, 256, 0, stream>>>(src, dst, n);
  };
  for (int s = 0; s < 2; s++) {
    cvt(wq[s], wqb[s], (size_t)D_ * D_);
    cvt(wk[s], wkb[s], (size_t)D_ * D_);
    cvt(wv[s], wvb[s], (size_t)D_ * D_);
    cvt(wo[s], wob[s], (size_t)D_ * D_);
    cvt(w1[s], w1b[s], (size_t)FF_ * D_);
    cvt(w2[s], w2b[s], (size_t)D_ * FF_);
    cvt(w3[s], w3b[s], (size_t)FF_ * D_);
  }

  rmsnorm_bf16_k<<<MROWS, 256, 0, stream>>>(x, attn_norm_w, xn);
  rmsnorm_bf16_k<<<MROWS, 256, 0, stream>>>(y, attn_norm_w, yn);

  const dim3 gD(D_ / 128, MROWS / 128);
  const dim3 gF(FF_ / 128, MROWS / 128);
  const dim3 gA(S_ / 128, H_, B_);

  for (int s = 0; s < 2; s++) {
    const __bf16* qin  = (s == 0) ? xn : yn;   // queries from own stream
    const __bf16* kvin = (s == 0) ? yn : xn;   // keys/values from the other
    const float*  resid = (s == 0) ? x : y;
    float* outp = (float*)d_out + (size_t)s * MROWS * D_;

    gemm_bf16_nt<<<gD, 256, 0, stream>>>(qin,  wqb[s], qb,  MROWS, D_, D_, 1);
    gemm_bf16_nt<<<gD, 256, 0, stream>>>(kvin, wkb[s], kb,  MROWS, D_, D_, 1);
    gemm_bf16_nt<<<gD, 256, 0, stream>>>(kvin, wvb[s], vb,  MROWS, D_, D_, 1);
    flash_attn_k<<<gA, 256, 0, stream>>>(qb, kb, vb, aob);
    gemm_bf16_nt<<<gD, 256, 0, stream>>>(aob, wob[s], hbf, MROWS, D_, D_, 1);
    gemm_bf16_nt<<<gF, 256, 0, stream>>>(hbf, w1b[s], g1b, MROWS, FF_, D_, 1);
    gemm_bf16_nt<<<gF, 256, 0, stream>>>(hbf, w3b[s], g3b, MROWS, FF_, D_, 1);
    swiglu_k<<<2048, 256, 0, stream>>>(g1b, g3b, ffb, (size_t)MROWS * FF_);
    gemm_bf16_nt<<<gD, 256, 0, stream>>>(ffb, w2b[s], fff, MROWS, D_, FF_, 0);
    resid_rmsnorm_k<<<MROWS, 256, 0, stream>>>(fff, resid, fnorm[s], outp);
  }
}
